// MultiHeadAttention_91311004713343
// MI455X (gfx1250) — matrix operand order repaired and checked by reading, not compiled, not run on hardware
//
#include <hip/hip_runtime.h>

// CDNA5 / gfx1250: wave32. Whole pipeline on v_wmma_f32_16x16x32_bf16 via
// bf16x3 splitting (hi*hi + hi*lo + lo*hi, fp32 accumulate). All float->bf16
// hi/lo splitting hoisted into bandwidth-bound pre-passes so the GEMM /
// attention inner loops are pure copy + ds_load_b128 + WMMA. (No explicit
// prefetch: the compiler's modulo scheduling already overlaps next-panel
// loads with WMMAs, and explicit prefetch was observed to break it.)
typedef __attribute__((ext_vector_type(8))) float v8f;
typedef __attribute__((ext_vector_type(8))) __bf16 v8bf;
typedef __attribute__((ext_vector_type(16))) __bf16 v16bf;

union BF16Frag {
  v16bf v;
  v8bf h[2];
  unsigned short s[16];
};

__device__ __forceinline__ v8f wmma_bf16(v16bf a, v16bf b, v8f c) {
#if defined(__gfx1250__)
  return __builtin_amdgcn_wmma_f32_16x16x32_bf16(false, a, false, b, (short)0,
                                                 c, false, false);
#else
  return c;
#endif
}

__device__ __forceinline__ unsigned short f2bf(float f) {  // RNE truncate
  unsigned int u = __float_as_uint(f);
  u += 0x7FFFu + ((u >> 16) & 1u);
  return (unsigned short)(u >> 16);
}
__device__ __forceinline__ float bf2f(unsigned short h) {
  return __uint_as_float(((unsigned int)h) << 16);
}

// ---------------------------------------------------------------------------
// Pre-pass: split fp32 array into bf16 hi + lo residual arrays.
// ---------------------------------------------------------------------------
__global__ __launch_bounds__(256) void split_bf16(
    const float* __restrict__ in, unsigned short* __restrict__ hi,
    unsigned short* __restrict__ lo, int n4) {
  int i = blockIdx.x * 256 + threadIdx.x;
  if (i >= n4) return;
  float4 v = ((const float4*)in)[i];
  ushort4 h, l;
  h.x = f2bf(v.x); l.x = f2bf(v.x - bf2f(h.x));
  h.y = f2bf(v.y); l.y = f2bf(v.y - bf2f(h.y));
  h.z = f2bf(v.z); l.z = f2bf(v.z - bf2f(h.z));
  h.w = f2bf(v.w); l.w = f2bf(v.w - bf2f(h.w));
  ((ushort4*)hi)[i] = h;
  ((ushort4*)lo)[i] = l;
}

// ---------------------------------------------------------------------------
// GEMM (bf16x3, pre-split operands): out[m,n] = sum_k A[m,k]*W[n,k] + bias[n]
// Block tile 128(M) x 64(N), K staged 32 wide (= one bf16 WMMA K).
// ---------------------------------------------------------------------------
#define ASTR 40
__global__ __launch_bounds__(256) void gemm_split_bf16x3(
    const unsigned short* __restrict__ Ah, const unsigned short* __restrict__ Al,
    const unsigned short* __restrict__ Wh, const unsigned short* __restrict__ Wl,
    const float* __restrict__ bias, float* __restrict__ out, int N, int K) {
  __shared__ unsigned short Ahi[128][ASTR], Alo[128][ASTR];
  __shared__ unsigned short Bhi[64][ASTR], Blo[64][ASTR];

  const int tid = threadIdx.x;
  const int lane = tid & 31;
  const int wave = tid >> 5;
  const int waveM = wave >> 1;
  const int waveN = wave & 1;
  const int l15 = lane & 15;
  const int mBase = blockIdx.y * 128;
  const int nBase = blockIdx.x * 64;

  v8f acc00 = {0, 0, 0, 0, 0, 0, 0, 0};
  v8f acc01 = acc00, acc10 = acc00, acc11 = acc00;

  const int mRow = waveM * 32 + l15;
  const int nRow = waveN * 32 + l15;
  const int lh8 = (lane >> 4) << 3;
  const int kb16 = (lane >> 4) << 4;

  for (int k0 = 0; k0 < K; k0 += 32) {
    __syncthreads();
    // A tile 128x32 ushorts = 512 uint4 per matrix; pure 16B copies.
#pragma unroll
    for (int p = 0; p < 2; ++p) {
      int idx = p * 256 + tid;
      int r = idx >> 2;
      int c8 = (idx & 3) << 3;
      *(uint4*)&Ahi[r][c8] =
          *(const uint4*)(Ah + (size_t)(mBase + r) * K + k0 + c8);
      *(uint4*)&Alo[r][c8] =
          *(const uint4*)(Al + (size_t)(mBase + r) * K + k0 + c8);
    }
    // B tile 64x32 ushorts = 256 uint4 per matrix.
    {
      int r = tid >> 2;
      int c8 = (tid & 3) << 3;
      *(uint4*)&Bhi[r][c8] =
          *(const uint4*)(Wh + (size_t)(nBase + r) * K + k0 + c8);
      *(uint4*)&Blo[r][c8] =
          *(const uint4*)(Wl + (size_t)(nBase + r) * K + k0 + c8);
    }
    __syncthreads();

    BF16Frag ah0, ah1, al0, al1, bh0, bh1, bl0, bl1;
    ah0.h[0] = *(const v8bf*)&Ahi[mRow][lh8];
    ah0.h[1] = *(const v8bf*)&Ahi[mRow][16 + lh8];
    al0.h[0] = *(const v8bf*)&Alo[mRow][lh8];
    al0.h[1] = *(const v8bf*)&Alo[mRow][16 + lh8];
    ah1.h[0] = *(const v8bf*)&Ahi[mRow + 16][lh8];
    ah1.h[1] = *(const v8bf*)&Ahi[mRow + 16][16 + lh8];
    al1.h[0] = *(const v8bf*)&Alo[mRow + 16][lh8];
    al1.h[1] = *(const v8bf*)&Alo[mRow + 16][16 + lh8];
    bh0.h[0] = *(const v8bf*)&Bhi[nRow][lh8];
    bh0.h[1] = *(const v8bf*)&Bhi[nRow][16 + lh8];
    bl0.h[0] = *(const v8bf*)&Blo[nRow][lh8];
    bl0.h[1] = *(const v8bf*)&Blo[nRow][16 + lh8];
    bh1.h[0] = *(const v8bf*)&Bhi[nRow + 16][lh8];
    bh1.h[1] = *(const v8bf*)&Bhi[nRow + 16][16 + lh8];
    bl1.h[0] = *(const v8bf*)&Blo[nRow + 16][lh8];
    bl1.h[1] = *(const v8bf*)&Blo[nRow + 16][16 + lh8];

    acc00 = wmma_bf16(ah0.v, bh0.v, acc00);
    acc00 = wmma_bf16(ah0.v, bl0.v, acc00);
    acc00 = wmma_bf16(al0.v, bh0.v, acc00);
    acc01 = wmma_bf16(ah0.v, bh1.v, acc01);
    acc01 = wmma_bf16(ah0.v, bl1.v, acc01);
    acc01 = wmma_bf16(al0.v, bh1.v, acc01);
    acc10 = wmma_bf16(ah1.v, bh0.v, acc10);
    acc10 = wmma_bf16(ah1.v, bl0.v, acc10);
    acc10 = wmma_bf16(al1.v, bh0.v, acc10);
    acc11 = wmma_bf16(ah1.v, bh1.v, acc11);
    acc11 = wmma_bf16(ah1.v, bl1.v, acc11);
    acc11 = wmma_bf16(al1.v, bh1.v, acc11);
  }

  const int rBase = (lane >> 4) << 3;
  const int col0 = nBase + waveN * 32 + l15;
  const int col1 = col0 + 16;
  const float bv0 = bias[col0];
  const float bv1 = bias[col1];
#pragma unroll
  for (int r = 0; r < 8; ++r) {
    int row0 = mBase + waveM * 32 + rBase + r;
    int row1 = row0 + 16;
    out[(size_t)row0 * N + col0] = acc00[r] + bv0;
    out[(size_t)row0 * N + col1] = acc01[r] + bv1;
    out[(size_t)row1 * N + col0] = acc10[r] + bv0;
    out[(size_t)row1 * N + col1] = acc11[r] + bv1;
  }
}

// ---------------------------------------------------------------------------
// Fused RoPE + split for q,k: qkv fp32 -> per-head [bh][t][d] bf16 hi/lo.
// ---------------------------------------------------------------------------
__global__ __launch_bounds__(256) void rope_split_qk(
    const float* __restrict__ qkv, const float* __restrict__ cs,
    const float* __restrict__ sn, unsigned short* __restrict__ qh,
    unsigned short* __restrict__ ql, unsigned short* __restrict__ kh,
    unsigned short* __restrict__ kl) {
  int gid = blockIdx.x * 256 + threadIdx.x;  // 4096*16*64 threads
  int d = gid & 63;
  int h = (gid >> 6) & 15;
  int m = gid >> 10;  // 0..4095
  int b = m >> 11;
  int t = m & 2047;
  int bh = b * 16 + h;
  size_t src = (size_t)m * 6144 + h * 128;
  float q1 = qkv[src + d], q2 = qkv[src + d + 64];
  float k1 = qkv[src + 2048 + d], k2 = qkv[src + 2048 + d + 64];
  float c1 = cs[t * 128 + d], s1 = sn[t * 128 + d];
  float c2 = cs[t * 128 + d + 64], s2 = sn[t * 128 + d + 64];
  float qa = q1 * c1 - q2 * s1;
  float qb = q2 * c2 + q1 * s2;
  float ka = k1 * c1 - k2 * s1;
  float kb = k2 * c2 + k1 * s2;
  size_t dst = ((size_t)bh * 2048 + t) * 128 + d;
  unsigned short x;
  x = f2bf(qa); qh[dst] = x;      ql[dst] = f2bf(qa - bf2f(x));
  x = f2bf(qb); qh[dst + 64] = x; ql[dst + 64] = f2bf(qb - bf2f(x));
  x = f2bf(ka); kh[dst] = x;      kl[dst] = f2bf(ka - bf2f(x));
  x = f2bf(kb); kh[dst + 64] = x; kl[dst + 64] = f2bf(kb - bf2f(x));
}

// ---------------------------------------------------------------------------
// V transpose + split: qkv v-part -> [bh][d][t] bf16 hi/lo, LDS-tiled.
// ---------------------------------------------------------------------------
__global__ __launch_bounds__(256) void v_split_t(
    const float* __restrict__ qkv, unsigned short* __restrict__ vth,
    unsigned short* __restrict__ vtl) {
  __shared__ float T[32][132];  // [t_local][d]
  const int tid = threadIdx.x;
  const int tt = blockIdx.x;   // 0..63
  const int bh = blockIdx.y;   // 0..31
  const int b = bh >> 4;
  const int h = bh & 15;
  const float* vsrc =
      qkv + (size_t)(b * 2048 + tt * 32) * 6144 + 4096 + h * 128;
#pragma unroll
  for (int p = 0; p < 4; ++p) {
    int idx = p * 256 + tid;
    int r = idx >> 5;          // t_local 0..31
    int c4 = (idx & 31) << 2;  // d
    *(float4*)&T[r][c4] = *(const float4*)(vsrc + (size_t)r * 6144 + c4);
  }
  __syncthreads();
  const int d = tid >> 1;         // 0..127
  const int th = (tid & 1) << 4;  // 0 or 16
  size_t drow = ((size_t)bh * 128 + d) * 2048 + tt * 32 + th;
#pragma unroll
  for (int i = 0; i < 4; ++i) {
    float f0 = T[th + i * 4 + 0][d];
    float f1 = T[th + i * 4 + 1][d];
    float f2 = T[th + i * 4 + 2][d];
    float f3 = T[th + i * 4 + 3][d];
    ushort4 hv, lv;
    hv.x = f2bf(f0); lv.x = f2bf(f0 - bf2f(hv.x));
    hv.y = f2bf(f1); lv.y = f2bf(f1 - bf2f(hv.y));
    hv.z = f2bf(f2); lv.z = f2bf(f2 - bf2f(hv.z));
    hv.w = f2bf(f3); lv.w = f2bf(f3 - bf2f(hv.w));
    *(ushort4*)(vth + drow + i * 4) = hv;
    *(ushort4*)(vtl + drow + i * 4) = lv;
  }
}

// ---------------------------------------------------------------------------
// Flash attention, fully bf16x3, all operands pre-split.
// Block = (b,h, 128-query tile); 8 waves x 16 query rows; 32-key tiles.
// Writes y pre-split (hi/lo) for the projection GEMM.
// ---------------------------------------------------------------------------
__global__ __launch_bounds__(256) void attn_flash(
    const unsigned short* __restrict__ qh, const unsigned short* __restrict__ ql,
    const unsigned short* __restrict__ kh, const unsigned short* __restrict__ kl,
    const unsigned short* __restrict__ vth,
    const unsigned short* __restrict__ vtl, unsigned short* __restrict__ yh,
    unsigned short* __restrict__ yl) {
  __shared__ unsigned short Khi[32][136], Klo[32][136];    // [key][d]
  __shared__ unsigned short Vthi[128][40], Vtlo[128][40];  // [d][key]
  __shared__ unsigned short Phi[8][16][40], Plo[8][16][40];

  const int tid = threadIdx.x;
  const int lane = tid & 31;
  const int wave = tid >> 5;
  const int l15 = lane & 15;
  const int lh8 = (lane >> 4) << 3;
  const int kb16 = (lane >> 4) << 4;
  const int qt = blockIdx.x;
  const int bh = blockIdx.y;
  const int b = bh >> 4;
  const int h = bh & 15;
  const int qRow = wave * 16 + l15;

  // Q fragments straight from pre-split global (16B loads, no conversion).
  const unsigned short* qrh = qh + ((size_t)bh * 2048 + qt * 128 + qRow) * 128;
  const unsigned short* qrl = ql + ((size_t)bh * 2048 + qt * 128 + qRow) * 128;
  BF16Frag qhi[4], qlo[4];
#pragma unroll
  for (int c = 0; c < 4; ++c) {
    qhi[c].h[0] = *(const v8bf*)(qrh + c * 32 + lh8);
    qhi[c].h[1] = *(const v8bf*)(qrh + c * 32 + 16 + lh8);
    qlo[c].h[0] = *(const v8bf*)(qrl + c * 32 + lh8);
    qlo[c].h[1] = *(const v8bf*)(qrl + c * 32 + 16 + lh8);
  }

  const v8f vzero = {0, 0, 0, 0, 0, 0, 0, 0};
  v8f o[8];
  float m[8], l[8];
#pragma unroll
  for (int n = 0; n < 8; ++n) o[n] = vzero;
#pragma unroll
  for (int r = 0; r < 8; ++r) {
    m[r] = -1e30f;
    l[r] = 0.0f;
  }

  for (int j = 0; j < 64; ++j) {  // 32 keys per step
    __syncthreads();
    // K tiles: 32x128 ushorts = 512 uint4 per matrix; pure copies.
    const unsigned short* kbh = kh + ((size_t)bh * 2048 + j * 32) * 128;
    const unsigned short* kbl = kl + ((size_t)bh * 2048 + j * 32) * 128;
#pragma unroll
    for (int p = 0; p < 2; ++p) {
      int idx = p * 256 + tid;
      int r = idx >> 4;
      int c8 = (idx & 15) << 3;
      *(uint4*)&Khi[r][c8] = *(const uint4*)(kbh + r * 128 + c8);
      *(uint4*)&Klo[r][c8] = *(const uint4*)(kbl + r * 128 + c8);
    }
    // V tiles (transposed layout): 128x32 ushorts = 512 uint4 per matrix.
    const unsigned short* vbh = vth + (size_t)bh * 128 * 2048 + j * 32;
    const unsigned short* vbl = vtl + (size_t)bh * 128 * 2048 + j * 32;
#pragma unroll
    for (int p = 0; p < 2; ++p) {
      int idx = p * 256 + tid;
      int r = idx >> 2;
      int c8 = (idx & 3) << 3;
      *(uint4*)&Vthi[r][c8] = *(const uint4*)(vbh + (size_t)r * 2048 + c8);
      *(uint4*)&Vtlo[r][c8] = *(const uint4*)(vbl + (size_t)r * 2048 + c8);
    }
    __syncthreads();

    // S = Q*K^T : two 16x16 tiles (keys 0..15, 16..31), 4 chunks x 3 terms.
    v8f s0 = vzero, s1 = vzero;
#pragma unroll
    for (int c = 0; c < 4; ++c) {
      int cb = c * 32 + lh8;
      BF16Frag bh0, bl0, bh1, bl1;
      bh0.h[0] = *(const v8bf*)&Khi[l15][cb];
      bh0.h[1] = *(const v8bf*)&Khi[l15][cb + 16];
      bl0.h[0] = *(const v8bf*)&Klo[l15][cb];
      bl0.h[1] = *(const v8bf*)&Klo[l15][cb + 16];
      bh1.h[0] = *(const v8bf*)&Khi[l15 + 16][cb];
      bh1.h[1] = *(const v8bf*)&Khi[l15 + 16][cb + 16];
      bl1.h[0] = *(const v8bf*)&Klo[l15 + 16][cb];
      bl1.h[1] = *(const v8bf*)&Klo[l15 + 16][cb + 16];
      s0 = wmma_bf16(qhi[c].v, bh0.v, s0);
      s0 = wmma_bf16(qhi[c].v, bl0.v, s0);
      s0 = wmma_bf16(qlo[c].v, bh0.v, s0);
      s1 = wmma_bf16(qhi[c].v, bh1.v, s1);
      s1 = wmma_bf16(qhi[c].v, bl1.v, s1);
      s1 = wmma_bf16(qlo[c].v, bh1.v, s1);
    }

    // online softmax over 32 keys
#pragma unroll
    for (int r = 0; r < 8; ++r) {
      float x0 = s0[r], x1 = s1[r];
      float mx = fmaxf(x0, x1);
      mx = fmaxf(mx, __shfl_xor(mx, 1));
      mx = fmaxf(mx, __shfl_xor(mx, 2));
      mx = fmaxf(mx, __shfl_xor(mx, 4));
      mx = fmaxf(mx, __shfl_xor(mx, 8));
      float mnew = fmaxf(m[r], mx);
      float alpha = exp2f(m[r] - mnew);
      float p0 = exp2f(x0 - mnew);
      float p1 = exp2f(x1 - mnew);
      float ps = p0 + p1;
      ps += __shfl_xor(ps, 1);
      ps += __shfl_xor(ps, 2);
      ps += __shfl_xor(ps, 4);
      ps += __shfl_xor(ps, 8);
      l[r] = l[r] * alpha + ps;
      m[r] = mnew;
#pragma unroll
      for (int n = 0; n < 8; ++n) o[n][r] *= alpha;
      int row = ((lane >> 4) << 3) + r;
      unsigned short h0 = f2bf(p0);
      Phi[wave][row][l15] = h0;
      Plo[wave][row][l15] = f2bf(p0 - bf2f(h0));
      unsigned short h1 = f2bf(p1);
      Phi[wave][row][l15 + 16] = h1;
      Plo[wave][row][l15 + 16] = f2bf(p1 - bf2f(h1));
    }
#if defined(__gfx1250__)
    asm volatile("s_wait_dscnt 0x0" ::: "memory");
#endif

    // O += P*V : contraction over 32 keys = one WMMA K; 8 d-tiles x 3 terms.
    BF16Frag aph, apl;
    aph.h[0] = *(const v8bf*)&Phi[wave][l15][lh8];
    aph.h[1] = *(const v8bf*)&Phi[wave][l15][16 + lh8];
    apl.h[0] = *(const v8bf*)&Plo[wave][l15][lh8];
    apl.h[1] = *(const v8bf*)&Plo[wave][l15][16 + lh8];
#pragma unroll
    for (int n = 0; n < 8; ++n) {
      int dr = n * 16 + l15;
      BF16Frag bvh, bvl;
      bvh.h[0] = *(const v8bf*)&Vthi[dr][lh8];
      bvh.h[1] = *(const v8bf*)&Vthi[dr][16 + lh8];
      bvl.h[0] = *(const v8bf*)&Vtlo[dr][lh8];
      bvl.h[1] = *(const v8bf*)&Vtlo[dr][16 + lh8];
      o[n] = wmma_bf16(aph.v, bvh.v, o[n]);
      o[n] = wmma_bf16(aph.v, bvl.v, o[n]);
      o[n] = wmma_bf16(apl.v, bvh.v, o[n]);
    }
  }

  // normalized output, written pre-split for the projection GEMM
#pragma unroll
  for (int r = 0; r < 8; ++r) {
    float inv = 1.0f / l[r];
    int trow = qt * 128 + wave * 16 + ((lane >> 4) << 3) + r;
    size_t orow = ((size_t)b * 2048 + trow) * 2048 + (size_t)h * 128;
#pragma unroll
    for (int n = 0; n < 8; ++n) {
      float val = o[n][r] * inv;
      unsigned short hb = f2bf(val);
      yh[orow + n * 16 + l15] = hb;
      yl[orow + n * 16 + l15] = f2bf(val - bf2f(hb));
    }
  }
}

// ---------------------------------------------------------------------------
extern "C" void kernel_launch(void* const* d_in, const int* in_sizes, int n_in,
                              void* d_out, int out_size, void* d_ws,
                              size_t ws_size, hipStream_t stream) {
  const float* x = (const float*)d_in[0];
  const float* cs = (const float*)d_in[1];
  const float* sn = (const float*)d_in[2];
  const float* Wqkv = (const float*)d_in[3];
  const float* bqkv = (const float*)d_in[4];
  const float* Wproj = (const float*)d_in[5];
  const float* bproj = (const float*)d_in[6];
  float* out = (float*)d_out;

  // workspace layout (all 16B aligned)
  char* w = (char*)d_ws;
  float* qkvBuf = (float*)w;  // 4096*6144*4 = 100663296 B
  unsigned short* qh = (unsigned short*)(w + 100663296);
  unsigned short* ql = qh + 8388608;
  unsigned short* kh = ql + 8388608;
  unsigned short* kl = kh + 8388608;
  unsigned short* vth = kl + 8388608;
  unsigned short* vtl = vth + 8388608;
  unsigned short* yh = vtl + 8388608;
  unsigned short* yl = yh + 8388608;
  unsigned short* ah = yl + 8388608;  // x split
  unsigned short* al = ah + 8388608;
  unsigned short* wh = al + 8388608;  // weight split (Wqkv, reused for Wproj)
  unsigned short* wl = wh + 12582912;

  split_bf16<<<8192, 256, 0, stream>>>(x, ah, al, 2097152);
  split_bf16<<<12288, 256, 0, stream>>>(Wqkv, wh, wl, 3145728);
  gemm_split_bf16x3<<<dim3(96, 32), 256, 0, stream>>>(ah, al, wh, wl, bqkv,
                                                      qkvBuf, 6144, 2048);
  rope_split_qk<<<16384, 256, 0, stream>>>(qkvBuf, cs, sn, qh, ql, kh, kl);
  v_split_t<<<dim3(64, 32), 256, 0, stream>>>(qkvBuf, vth, vtl);
  attn_flash<<<dim3(16, 32), 256, 0, stream>>>(qh, ql, kh, kl, vth, vtl, yh,
                                               yl);
  split_bf16<<<4096, 256, 0, stream>>>(Wproj, wh, wl, 1048576);
  gemm_split_bf16x3<<<dim3(32, 32), 256, 0, stream>>>(yh, yl, wh, wl, bproj,
                                                      out, 2048, 2048);
}
